// MultiHeadAttention_38671885533756
// MI455X (gfx1250) — matrix operand order repaired and checked by reading, not compiled, not run on hardware
//
#include <hip/hip_runtime.h>
#include <hip/hip_bf16.h>

// ---------------------------------------------------------------------------
// MI455X (gfx1250) fused multi-head attention forward.
//   B=4, S=1024, D=1024, H=16, HD=64
//   d_out = [ out (4*1024*1024 f32) | attn (4*16*1024*1024 f32) ]
// Pipeline:
//   0) Wq/Wk/Wv/Wo f32 -> f16 (one bandwidth-trivial pass; weights are then
//      async-copied global->LDS with no VGPR pass-through)
//   1) Q = query@Wq^T+bq  -> f16 [B,H,S,HD]      (WMMA GEMM, LDS-staged W)
//   2) K = key  @Wk^T+bk  -> f16 [B,H,S,HD]
//   3) Vt= value@Wv^T+bv  -> f16 [B,H,HD,S]  (transposed for WMMA B-frags)
//   4) attention: per (b,h,64-row q-strip): scores in 256KB LDS, softmax,
//      write attn to d_out once, ctx = attn@V via WMMA, ctx -> f16 [B,S,D]
//   5) out = ctx@Wo^T+bo -> f32 d_out            (WMMA GEMM, LDS-staged W)
// ---------------------------------------------------------------------------

typedef __attribute__((ext_vector_type(16))) _Float16 v16h;
typedef __attribute__((ext_vector_type(8)))  _Float16 v8h;
typedef __attribute__((ext_vector_type(4)))  _Float16 v4h;
typedef __attribute__((ext_vector_type(8)))  float    v8f;
typedef __attribute__((ext_vector_type(4)))  float    v4f;

union FragH { v16h v; v8h h[2]; };

// ---- CDNA5 async global->LDS copy (ASYNCcnt path, CDNA5 ISA) -----
__device__ inline void async_copy_b128(unsigned lds_addr, const void* gaddr) {
    asm volatile("global_load_async_to_lds_b128 %0, %1, off"
                 :: "v"(lds_addr), "v"((unsigned long long)(uintptr_t)gaddr)
                 : "memory");
}
__device__ inline void wait_async0() {
    asm volatile("s_wait_asynccnt 0x0" ::: "memory");
}

// ---- WMMA fragment loaders (wave32 layouts, CDNA5 ISA 7.12.2) --------
// A-matrix 16x32 f16: lane m=lane&15, half=lane>>4.
//   a[0..7]  <-> k = half*8 + 0..7        (16B contiguous)
//   a[8..15] <-> k = 16 + half*8 + 0..7   (16B contiguous)
__device__ inline v16h load_a_f16(const _Float16* base /*(row0,k0)*/, int ld, int lane) {
    int m = lane & 15, half = lane >> 4;
    const _Float16* p = base + (size_t)m * ld;
    FragH f;
    f.h[0] = *(const v8h*)(p + half * 8);
    f.h[1] = *(const v8h*)(p + 16 + half * 8);
    return f.v;
}

__device__ inline v16h load_a_f32(const float* base /*(row0,k0)*/, int ld, int lane) {
    int m = lane & 15, half = lane >> 4;
    const float* p = base + (size_t)m * ld + half * 8;
    v4f x0 = *(const v4f*)(p + 0);
    v4f x1 = *(const v4f*)(p + 4);
    v4f x2 = *(const v4f*)(p + 16);
    v4f x3 = *(const v4f*)(p + 20);
    FragH f;
#pragma unroll
    for (int j = 0; j < 4; ++j) {
        f.h[0][j]     = (_Float16)x0[j];
        f.h[0][4 + j] = (_Float16)x1[j];
        f.h[1][j]     = (_Float16)x2[j];
        f.h[1][4 + j] = (_Float16)x3[j];
    }
    return f.v;
}

// B-matrix 32x16 f16 loaded from a [n][k] row-major source (i.e. B^T):
//   lane n=lane&15, half=lane>>4; b[0..7] <-> k = half*8 + 0..7, b[8..15] <-> k = 16 + half*8 + 0..7
__device__ inline v16h load_b_f16(const _Float16* base /*(n0,k0)*/, int ld, int lane) {
    int n = lane & 15, half = lane >> 4;
    const _Float16* p = base + (size_t)n * ld;
    FragH f;
    f.h[0] = *(const v8h*)(p + half * 8);
    f.h[1] = *(const v8h*)(p + 16 + half * 8);
    return f.v;
}

__device__ inline v8f wmma32(v16h a, v16h b, v8f c) {
    // v_wmma_f32_16x16x32_f16, f32 accumulate
    return __builtin_amdgcn_wmma_f32_16x16x32_f16(
        /*neg_a=*/false, a, /*neg_b=*/false, b,
        /*c_mod=*/(short)0, c, /*reuse_a=*/false, /*reuse_b=*/false);
}

// ---------------------------------------------------------------------------
// Weight pre-conversion: f32 -> f16, vectorized 4-wide.
// ---------------------------------------------------------------------------
__global__ __launch_bounds__(256) void f32_to_f16_kernel(
    const float* __restrict__ src, _Float16* __restrict__ dst, int n4)
{
    const int i = blockIdx.x * 256 + threadIdx.x;
    if (i < n4) {
        v4f x = ((const v4f*)src)[i];
        v4h y;
#pragma unroll
        for (int j = 0; j < 4; ++j) y[j] = (_Float16)x[j];
        ((v4h*)dst)[i] = y;
    }
}

// ---------------------------------------------------------------------------
// Projection GEMM: C[m,n] = sum_k A[m,k] * Wh[n,k] + bias[n]   (Wh = f16)
//   AMODE: 0 = f32 A source, 1 = f16 A source
//   OMODE: 0 = f16 out [B,H,S,HD], 1 = f16 out [B,H,HD,S] (V-transposed),
//          2 = f32 out [M,N]
// Block: 256 thr (8 waves). Tile: 128 rows (16/wave) x 128 cols (8 C-frags).
// W slab (128 cols x 128 k, f16) is async-copied global->LDS once per block
// (vs 8x redundant per-wave global loads), double-buffered behind the WMMAs.
// Slab rows padded to 136 f16 (68 dwords) so the 16 fragment rows map to
// distinct LDS bank groups.
// ---------------------------------------------------------------------------
template <int AMODE, int OMODE>
__global__ __launch_bounds__(256) void proj_gemm_kernel(
    const void* __restrict__ Aptr, const _Float16* __restrict__ Wh,
    const float* __restrict__ bias, void* __restrict__ Optr)
{
    constexpr int K = 1024, N = 1024;
    constexpr int LDW = 136;                    // 128 + 8 f16 pad
    __shared__ _Float16 sW[2][128 * LDW];       // 2 x 34 KB

    const int tid = threadIdx.x, w = tid >> 5, lane = tid & 31;
    const int row0 = blockIdx.y * 128 + w * 16;
    const int col0 = blockIdx.x * 128;

    // Cooperative async stage: thread -> (slab row cr, 128B half-row ch),
    // 8 x b128 = 128B per thread, 32KB per stage.
    const int cr = tid >> 1;
    const int ch = tid & 1;

    auto issue_stage = [&](int ks, int buf) {
        const _Float16* gsrc = Wh + (size_t)(col0 + cr) * K + ks * 128 + ch * 64;
        unsigned laddr = (unsigned)(uintptr_t)(&sW[buf][cr * LDW + ch * 64]);
#pragma unroll
        for (int j = 0; j < 8; ++j)
            async_copy_b128(laddr + j * 16, gsrc + j * 8);
    };

    v8f acc[8] = {};
    issue_stage(0, 0);

    for (int ks = 0; ks < 8; ++ks) {            // 8 slabs of 128 k
        wait_async0();                          // own copies done
        __syncthreads();                        // everyone's copies done
        if (ks < 7) issue_stage(ks + 1, (ks + 1) & 1);
        const _Float16* wbuf = sW[ks & 1];

#pragma unroll
        for (int j = 0; j < 4; ++j) {           // 4 k-steps of 32 per slab
            const int kk = ks * 128 + j * 32;
            v16h a;
            if (AMODE == 0)
                a = load_a_f32((const float*)Aptr + (size_t)row0 * K + kk, K, lane);
            else
                a = load_a_f16((const _Float16*)Aptr + (size_t)row0 * K + kk, K, lane);
#pragma unroll
            for (int nt = 0; nt < 8; ++nt) {
                v16h bf = load_b_f16(wbuf + (size_t)(nt * 16) * LDW + j * 32, LDW, lane);
                acc[nt] = wmma32(a, bf, acc[nt]);
            }
        }
        __syncthreads();                        // all waves done with this buf
    }

    const int nl = lane & 15, half = lane >> 4;
#pragma unroll
    for (int nt = 0; nt < 8; ++nt) {
        const int n  = col0 + nt * 16 + nl;
        const float bv = bias[n];
#pragma unroll
        for (int r = 0; r < 8; ++r) {
            const int m = row0 + r + half * 8;        // C/D: m = r (+8 upper half)
            const float val = acc[nt][r] + bv;
            if (OMODE == 2) {
                ((float*)Optr)[(size_t)m * N + n] = val;
            } else {
                const int bb = m >> 10, s  = m & 1023;
                const int hh = n >> 6,  hd = n & 63;
                const _Float16 hv = (_Float16)val;
                if (OMODE == 0)
                    ((_Float16*)Optr)[(((size_t)(bb * 16 + hh) * 1024 + s) * 64 + hd)] = hv;
                else
                    ((_Float16*)Optr)[(((size_t)(bb * 16 + hh) * 64 + hd) * 1024 + s)] = hv;
            }
        }
    }
}

// ---------------------------------------------------------------------------
// Attention: one block per (b, h, 64-row q-strip).
// LDS: 64x1024 f32 score panel (256 KB, CDNA5 320KB WGP LDS) + row stats.
// ---------------------------------------------------------------------------
__global__ __launch_bounds__(256) void attention_kernel(
    const _Float16* __restrict__ Q,   // [B,H,S,HD] f16
    const _Float16* __restrict__ Kx,  // [B,H,S,HD] f16
    const _Float16* __restrict__ Vt,  // [B,H,HD,S] f16
    const int* __restrict__ mask,     // [B,1,1,S]
    float* __restrict__ attn,         // [B,H,S,S] f32 (part of d_out)
    _Float16* __restrict__ ctx)       // [B,S,D]  f16
{
    extern __shared__ char smem[];
    float* sS      = (float*)smem;          // [64][1024]
    float* rowmax  = sS + 64 * 1024;        // [64]
    float* rowrinv = rowmax + 64;           // [64]

    const int b = blockIdx.z, h = blockIdx.y, qt = blockIdx.x;
    const int tid = threadIdx.x, w = tid >> 5, lane = tid & 31;
    const int nl = lane & 15, half = lane >> 4;

    const _Float16* Qh = Q  + ((size_t)(b * 16 + h) * 1024 + qt * 64) * 64;
    const _Float16* Kh = Kx + ((size_t)(b * 16 + h) * 1024) * 64;
    const _Float16* Vh = Vt + ((size_t)(b * 16 + h) * 64) * 1024;

    // ---- Phase 1: scores = (Q K^T)/8, masked, into LDS -------------------
    const int mt = w >> 1, nh = w & 1;
    const v16h aq0 = load_a_f16(Qh + (size_t)mt * 16 * 64 + 0,  64, lane);
    const v16h aq1 = load_a_f16(Qh + (size_t)mt * 16 * 64 + 32, 64, lane);

    for (int nt = nh * 32; nt < nh * 32 + 32; ++nt) {
        v16h b0 = load_b_f16(Kh + (size_t)nt * 16 * 64 + 0,  64, lane);
        v16h b1 = load_b_f16(Kh + (size_t)nt * 16 * 64 + 32, 64, lane);
        v8f c = {};
        c = wmma32(aq0, b0, c);
        c = wmma32(aq1, b1, c);
        const int ncol = nt * 16 + nl;
        const bool keep = (mask[b * 1024 + ncol] != 0);
        const int mbase = mt * 16 + half * 8;
#pragma unroll
        for (int r = 0; r < 8; ++r) {
            const float sv = c[r] * 0.125f;           // 1/sqrt(64)
            sS[(size_t)(mbase + r) * 1024 + ncol] = keep ? sv : -1e10f;
        }
    }
    __syncthreads();

    // ---- Phase 2: softmax row stats (wave w owns rows w*8 .. w*8+7) -------
    for (int rr = w * 8; rr < w * 8 + 8; ++rr) {
        const float* row = sS + (size_t)rr * 1024;
        float mx = -3.4e38f;
        for (int j = lane; j < 1024; j += 32) mx = fmaxf(mx, row[j]);
#pragma unroll
        for (int off = 16; off > 0; off >>= 1) mx = fmaxf(mx, __shfl_xor(mx, off, 32));
        float s = 0.f;
        for (int j = lane; j < 1024; j += 32) s += __expf(row[j] - mx);
#pragma unroll
        for (int off = 16; off > 0; off >>= 1) s += __shfl_xor(s, off, 32);
        if (lane == 0) { rowmax[rr] = mx; rowrinv[rr] = 1.0f / s; }
    }
    __syncthreads();

    // ---- Phase 3: normalize in LDS + single coalesced write of attn -------
    float* attnOut = attn + ((size_t)(b * 16 + h) * 1024 + qt * 64) * 1024;
    for (int e = tid; e < 64 * 1024; e += 256) {
        const int r = e >> 10;
        const float p = __expf(sS[e] - rowmax[r]) * rowrinv[r];
        sS[e] = p;                // reuse as A-operand of attn @ V
        attnOut[e] = p;
    }
    __syncthreads();

    // ---- Phase 4: ctx = attn @ V  (A from LDS, B from transposed V) -------
    v8f c0 = {}, c1 = {};
    for (int kk = 0; kk < 1024; kk += 32) {
        v16h ap = load_a_f32(sS + (size_t)mt * 16 * 1024 + kk, 1024, lane);
        v16h v0 = load_b_f16(Vh + (size_t)(nh * 32 + 0)  * 1024 + kk, 1024, lane);
        v16h v1 = load_b_f16(Vh + (size_t)(nh * 32 + 16) * 1024 + kk, 1024, lane);
        c0 = wmma32(ap, v0, c0);
        c1 = wmma32(ap, v1, c1);
    }
    _Float16* cbase = ctx + ((size_t)(b * 1024 + qt * 64)) * 1024 + h * 64;
#pragma unroll
    for (int r = 0; r < 8; ++r) {
        const int m = mt * 16 + half * 8 + r;
        cbase[(size_t)m * 1024 + nh * 32 + nl]      = (_Float16)c0[r];
        cbase[(size_t)m * 1024 + nh * 32 + 16 + nl] = (_Float16)c1[r];
    }
}

// ---------------------------------------------------------------------------
extern "C" void kernel_launch(void* const* d_in, const int* in_sizes, int n_in,
                              void* d_out, int out_size, void* d_ws, size_t ws_size,
                              hipStream_t stream) {
    (void)in_sizes; (void)n_in; (void)out_size; (void)ws_size;

    const float* query = (const float*)d_in[0];
    const float* key   = (const float*)d_in[1];
    const float* value = (const float*)d_in[2];
    const int*   mask  = (const int*)d_in[3];
    const float* Wq = (const float*)d_in[4];  const float* bq = (const float*)d_in[5];
    const float* Wk = (const float*)d_in[6];  const float* bk = (const float*)d_in[7];
    const float* Wv = (const float*)d_in[8];  const float* bv = (const float*)d_in[9];
    const float* Wo = (const float*)d_in[10]; const float* bo = (const float*)d_in[11];

    float* out  = (float*)d_out;                          // [4,1024,1024]
    float* attn = out + (size_t)4 * 1024 * 1024;          // [4,16,1024,1024]

    const size_t PROJ_ELEMS = (size_t)4 * 16 * 1024 * 64; // 4 Mi f16 each
    const size_t W_ELEMS    = (size_t)1024 * 1024;        // 1 Mi f16 each
    _Float16* Qh  = (_Float16*)d_ws;
    _Float16* Kh  = Qh  + PROJ_ELEMS;
    _Float16* Vt  = Kh  + PROJ_ELEMS;
    _Float16* ctx = Vt  + PROJ_ELEMS;                     // [B,S,D] f16
    _Float16* Wqh = ctx + PROJ_ELEMS;
    _Float16* Wkh = Wqh + W_ELEMS;
    _Float16* Wvh = Wkh + W_ELEMS;
    _Float16* Woh = Wvh + W_ELEMS;

    const int SMEM = 64 * 1024 * 4 + 128 * 4;             // 256KB panel + stats
    hipFuncSetAttribute(reinterpret_cast<const void*>(attention_kernel),
                        hipFuncAttributeMaxDynamicSharedMemorySize, SMEM);

    dim3 blk(256);
    // Weight pre-conversion: 4 x 1Mi f32 -> f16 (n4 = 256Ki float4 each)
    const int n4 = (int)(W_ELEMS / 4);
    f32_to_f16_kernel<<<dim3(n4 / 256), blk, 0, stream>>>(Wq, Wqh, n4);
    f32_to_f16_kernel<<<dim3(n4 / 256), blk, 0, stream>>>(Wk, Wkh, n4);
    f32_to_f16_kernel<<<dim3(n4 / 256), blk, 0, stream>>>(Wv, Wvh, n4);
    f32_to_f16_kernel<<<dim3(n4 / 256), blk, 0, stream>>>(Wo, Woh, n4);

    dim3 pgrid(8, 32);
    proj_gemm_kernel<0, 0><<<pgrid, blk, 0, stream>>>((const void*)query, Wqh, bq, (void*)Qh);
    proj_gemm_kernel<0, 0><<<pgrid, blk, 0, stream>>>((const void*)key,   Wkh, bk, (void*)Kh);
    proj_gemm_kernel<0, 1><<<pgrid, blk, 0, stream>>>((const void*)value, Wvh, bv, (void*)Vt);

    attention_kernel<<<dim3(16, 16, 4), blk, SMEM, stream>>>(Qh, Kh, Vt, mask, attn, ctx);

    proj_gemm_kernel<1, 2><<<pgrid, blk, 0, stream>>>((const void*)ctx, Woh, bo, (void*)out);
}
